// CFNO_84430467105059
// MI455X (gfx1250) — compile-verified, matrix operand order checked by reading, not run on hardware
//
#include <hip/hip_runtime.h>
#include <hip/hip_bf16.h>
#include <math.h>

typedef __attribute__((ext_vector_type(16))) _Float16 v16h;
typedef __attribute__((ext_vector_type(8)))  float    v8f;
typedef __attribute__((ext_vector_type(4)))  float    vf4;

// ---- problem constants ----
#define HW      2048
#define PH      128          // patch grid 128x128
#define BATCH   8
#define DIN     256
#define DOUT    16
#define NGRP    8
#define CNT_PER_GROUP 32768.0f   // 2 channels * 128*128
#define TWO_PI  6.28318530717958647692f

// ---- workspace layout (floats) ----
#define OFF_S     0            // 16*256 complex  = 8192 f
#define OFF_T     8192         // 16*256          = 4096 f
#define OFF_M     12288        // 16*256          = 4096 f
#define OFF_V     16384        // 16 (padded)     = 32 f
#define OFF_PS    16416        // 1024 blocks * 8 = 8192 f
#define OFF_PSS   24608        // 1024 blocks * 8 = 8192 f
#define OFF_STATS 32800        // 64*(mean,rstd)  = 128 f
// total ~ 32928 floats = ~132 KB

// ---------------- K1: S[e][k] = sum_d e^{+2pi i e d/16} * Wc[d][k] ----------------
__global__ void cfno_k_S(const float* __restrict__ wr, const float* __restrict__ wi,
                         float* __restrict__ ws) {
    int e = blockIdx.x;     // 0..15
    int k = threadIdx.x;    // 0..255
    float Sr = 0.f, Si = 0.f;
    for (int d = 0; d < 16; ++d) {
        float ang = (float)((e * d) & 15) * (TWO_PI / 16.f);
        float sn, cs;
        __sincosf(ang, &sn, &cs);
        float a = wr[d * DIN + k], b = wi[d * DIN + k];
        Sr += a * cs - b * sn;
        Si += a * sn + b * cs;
    }
    float* S = ws + OFF_S;
    S[(e * DIN + k) * 2 + 0] = Sr;
    S[(e * DIN + k) * 2 + 1] = Si;
}

// ---------------- K2: T[e][n] = (1/64) * Re( sum_k S[e][k] e^{-2pi i k n/256} ) ----------------
__global__ void cfno_k_T(float* __restrict__ ws) {
    int e = blockIdx.x;     // 0..15
    int n = threadIdx.x;    // 0..255
    const float* S = ws + OFF_S + e * (DIN * 2);
    float acc = 0.f;
    for (int k = 0; k < DIN; ++k) {
        float phi = (float)((k * n) & 255) * (TWO_PI / 256.f);
        float sn, cs;
        __sincosf(phi, &sn, &cs);
        acc += S[2 * k] * cs + S[2 * k + 1] * sn;  // Re(S * (cos - i sin))
    }
    ws[OFF_T + e * DIN + n] = acc * (1.f / 64.f);
}

// ---------------- K3: M[o][n] = sum_e conv_w[o][e] T[e][n] ; v[o] ----------------
__global__ void cfno_k_M(const float* __restrict__ cw, const float* __restrict__ cb,
                         const float* __restrict__ br, const float* __restrict__ bi,
                         float* __restrict__ ws) {
    int n = threadIdx.x;    // 0..255
    const float* T = ws + OFF_T;
    for (int o = 0; o < 16; ++o) {
        float acc = 0.f;
        for (int e = 0; e < 16; ++e) acc += cw[o * 16 + e] * T[e * DIN + n];
        ws[OFF_M + o * DIN + n] = acc;
    }
    if (n < 16) {
        int o = n;
        float acc = cb[o];
        for (int e = 0; e < 16; ++e) {
            float u = 0.f;
            for (int d = 0; d < 16; ++d) {
                float ang = (float)((e * d) & 15) * (TWO_PI / 16.f);
                float sn, cs;
                __sincosf(ang, &sn, &cs);
                u += br[d] * cs - bi[d] * sn;     // Re((br+i bi)(cos+i sin))
            }
            acc += cw[o * 16 + e] * (u * 0.25f);
        }
        ws[OFF_V + o] = acc;
    }
}

// ---------------- K4: main GEMM via v_wmma_f32_16x16x32_f16 ----------------
// grid: (128 ph, 8 b), block 256 = 8 waves; wave w handles patches pw0 = 16*w .. +15.
// A row (patch) = 16x16 tile of x; K index k -> x[ph*16 + k/16][pw*16 + k%16].
__global__ __launch_bounds__(256) void cfno_k_gemm(const float* __restrict__ x,
                                                   const float* __restrict__ ws,
                                                   float* __restrict__ out,
                                                   float* __restrict__ ps,
                                                   float* __restrict__ pss) {
    const int ph   = blockIdx.x;
    const int b    = blockIdx.y;
    const int tid  = threadIdx.x;
    const int wave = tid >> 5;
    const int lane = tid & 31;
    const int hi   = lane >> 4;     // 0: K%16 in [0,8), 1: in [8,16)
    const int lo   = lane & 15;     // patch-in-tile (A) or column/channel (B/D)
    const int pw0  = wave * 16;

    // ---- B fragments: B[k][n] = M[n][k], ISA 16-bit layout ----
    // element e of chunk c maps to K = 32c + (e&7) + 16*(e>>3) + 8*hi, column n = lo
    v16h Bf[8];
    {
        const float* mrow = ws + OFF_M + lo * DIN + hi * 8;
#pragma unroll
        for (int c = 0; c < 8; ++c) {
            const float* p0 = mrow + 32 * c;
            const float* p1 = p0 + 16;
            vf4 b0 = *(const vf4*)(p0);
            vf4 b1 = *(const vf4*)(p0 + 4);
            vf4 b2 = *(const vf4*)(p1);
            vf4 b3 = *(const vf4*)(p1 + 4);
            v16h t;
            t[0]  = (_Float16)b0.x; t[1]  = (_Float16)b0.y;
            t[2]  = (_Float16)b0.z; t[3]  = (_Float16)b0.w;
            t[4]  = (_Float16)b1.x; t[5]  = (_Float16)b1.y;
            t[6]  = (_Float16)b1.z; t[7]  = (_Float16)b1.w;
            t[8]  = (_Float16)b2.x; t[9]  = (_Float16)b2.y;
            t[10] = (_Float16)b2.z; t[11] = (_Float16)b2.w;
            t[12] = (_Float16)b3.x; t[13] = (_Float16)b3.y;
            t[14] = (_Float16)b3.z; t[15] = (_Float16)b3.w;
            Bf[c] = t;
        }
    }

    // ---- A fragments straight from x + chained WMMA over K=256 ----
    v8f acc = {};
    const float* xb = x + ((size_t)b << 22)                        // b*2048*2048
                        + (size_t)(ph * 16) * HW
                        + (size_t)(pw0 + lo) * 16 + hi * 8;
#pragma unroll
    for (int c = 0; c < 8; ++c) {
        const float* r0 = xb + (size_t)(2 * c) * HW;   // K rows 2c, 2c+1 of the patch
        const float* r1 = r0 + HW;
        vf4 a0 = *(const vf4*)(r0);
        vf4 a1 = *(const vf4*)(r0 + 4);
        vf4 a2 = *(const vf4*)(r1);
        vf4 a3 = *(const vf4*)(r1 + 4);
        v16h Af;
        Af[0]  = (_Float16)a0.x; Af[1]  = (_Float16)a0.y;
        Af[2]  = (_Float16)a0.z; Af[3]  = (_Float16)a0.w;
        Af[4]  = (_Float16)a1.x; Af[5]  = (_Float16)a1.y;
        Af[6]  = (_Float16)a1.z; Af[7]  = (_Float16)a1.w;
        Af[8]  = (_Float16)a2.x; Af[9]  = (_Float16)a2.y;
        Af[10] = (_Float16)a2.z; Af[11] = (_Float16)a2.w;
        Af[12] = (_Float16)a3.x; Af[13] = (_Float16)a3.y;
        Af[14] = (_Float16)a3.z; Af[15] = (_Float16)a3.w;
        acc = __builtin_amdgcn_wmma_f32_16x16x32_f16(
            /*neg_a=*/false, Af, /*neg_b=*/false, Bf[c],
            /*c_mod=*/(short)0, acc, /*reuse_a=*/false, /*reuse_b=*/false);
    }

    // ---- bias (folded FFT/linear/conv bias), D layout: vgpr r -> patch (hi*8+r), lane -> channel lo
    const float bias = ws[OFF_V + lo];
#pragma unroll
    for (int r = 0; r < 8; ++r) acc[r] += bias;

    size_t obase = ((((size_t)(b * DOUT + lo)) * PH + ph) * PH) + pw0 + hi * 8;
    vf4 s0 = { acc[0], acc[1], acc[2], acc[3] };
    vf4 s1 = { acc[4], acc[5], acc[6], acc[7] };
    *(vf4*)(out + obase)     = s0;
    *(vf4*)(out + obase + 4) = s1;

    // ---- deterministic GroupNorm partials (fixed-order block reduction, no atomics) ----
    float s = 0.f, ss = 0.f;
#pragma unroll
    for (int r = 0; r < 8; ++r) { s += acc[r]; ss += acc[r] * acc[r]; }
    __shared__ float lsum[256];
    __shared__ float lss[256];
    lsum[tid] = s; lss[tid] = ss;
    __syncthreads();
    if (tid < NGRP) {
        int g = tid;
        float S = 0.f, SS = 0.f;
        for (int w = 0; w < 8; ++w)
            for (int h = 0; h < 2; ++h)
                for (int dd = 0; dd < 2; ++dd) {
                    int t2 = w * 32 + h * 16 + 2 * g + dd;
                    S += lsum[t2]; SS += lss[t2];
                }
        int blk = b * PH + ph;
        ps[blk * NGRP + g]  = S;
        pss[blk * NGRP + g] = SS;
    }
}

// ---------------- K5: final stats per (b, group) ----------------
__global__ void cfno_k_stats(const float* __restrict__ ps, const float* __restrict__ pss,
                             float* __restrict__ stats) {
    int t = threadIdx.x;            // 0..63
    if (t >= BATCH * NGRP) return;
    int b = t >> 3, g = t & 7;
    float S = 0.f, SS = 0.f;
    for (int p = 0; p < PH; ++p) {
        int blk = b * PH + p;
        S  += ps[blk * NGRP + g];
        SS += pss[blk * NGRP + g];
    }
    float mean = S / CNT_PER_GROUP;
    float var  = SS / CNT_PER_GROUP - mean * mean;
    stats[t * 2 + 0] = mean;
    stats[t * 2 + 1] = rsqrtf(var + 1e-5f);
}

// ---------------- K6: normalize in place (float4 vectorized) ----------------
__global__ __launch_bounds__(256) void cfno_k_norm(float* __restrict__ out,
                                                   const float* __restrict__ stats,
                                                   const float* __restrict__ gamma,
                                                   const float* __restrict__ beta) {
    size_t i = ((size_t)blockIdx.x * 256 + threadIdx.x) * 4;
    int bd = (int)(i >> 14);        // 128*128 = 16384 spatial per (b,d)
    int d = bd & 15, b = bd >> 4;
    int g = d >> 1;
    float mean = stats[(b * NGRP + g) * 2 + 0];
    float rstd = stats[(b * NGRP + g) * 2 + 1];
    float sc = rstd * gamma[d];
    float sh = beta[d] - mean * sc;
    vf4 v = *(vf4*)(out + i);
    v = v * sc + sh;
    *(vf4*)(out + i) = v;
}

extern "C" void kernel_launch(void* const* d_in, const int* in_sizes, int n_in,
                              void* d_out, int out_size, void* d_ws, size_t ws_size,
                              hipStream_t stream) {
    const float* x     = (const float*)d_in[0];
    const float* fc_wr = (const float*)d_in[1];
    const float* fc_wi = (const float*)d_in[2];
    const float* fc_br = (const float*)d_in[3];
    const float* fc_bi = (const float*)d_in[4];
    const float* cw    = (const float*)d_in[5];
    const float* cb    = (const float*)d_in[6];
    const float* gamma = (const float*)d_in[7];
    const float* beta  = (const float*)d_in[8];
    float* out = (float*)d_out;
    float* ws  = (float*)d_ws;

    cfno_k_S<<<16, 256, 0, stream>>>(fc_wr, fc_wi, ws);
    cfno_k_T<<<16, 256, 0, stream>>>(ws);
    cfno_k_M<<<1, 256, 0, stream>>>(cw, cb, fc_br, fc_bi, ws);
    cfno_k_gemm<<<dim3(PH, BATCH), 256, 0, stream>>>(x, ws, out,
                                                     ws + OFF_PS, ws + OFF_PSS);
    cfno_k_stats<<<1, 64, 0, stream>>>(ws + OFF_PS, ws + OFF_PSS, ws + OFF_STATS);
    cfno_k_norm<<<(out_size / 4 + 255) / 256, 256, 0, stream>>>(out, ws + OFF_STATS,
                                                                gamma, beta);
}
